// LSTMAggregator_67757404061982
// MI455X (gfx1250) — matrix operand order repaired and checked by reading, not compiled, not run on hardware
//
#include <hip/hip_runtime.h>

// ---------------- problem constants (match reference) ----------------
#define V_  100000
#define D_  256
#define N_  8192
#define S_  20
#define H_  256
#define GDIM 1024          // 4*H
#define KTOT 512           // D + H (fused [x | h] GEMM K)

// ---------------- blocking ----------------
#define NB  64             // nodes per workgroup (M blocking -> weight reuse)
#define NW  16             // waves per workgroup
#define WG_THREADS (NW * 32)

#define WPACK_PER_DIR (KTOT * GDIM)   // 524288 bf16 per direction

typedef __attribute__((ext_vector_type(16))) __bf16 v16bf;
typedef __attribute__((ext_vector_type(8)))  float  v8f;

// ---------------------------------------------------------------------
// Weight packing: combined B matrix B[k,n] (k<256 -> w_ih[n,k], else w_hh[n,k-256])
// stored per 32x16 (KxN) WMMA-B fragment, lane-linear:
//   lane = n_in + 16*bit3(k_in), slot = (k_in&7) + 8*bit4(k_in), 16 bf16 (32B) contiguous per lane.
// ---------------------------------------------------------------------
__global__ void pack_weights_kernel(const float* __restrict__ w_ih_f,
                                    const float* __restrict__ w_hh_f,
                                    const float* __restrict__ w_ih_b,
                                    const float* __restrict__ w_hh_b,
                                    __bf16* __restrict__ wpack) {
  int gid = blockIdx.x * blockDim.x + threadIdx.x;
  if (gid >= 2 * WPACK_PER_DIR) return;
  int d   = gid >> 19;                 // direction
  int rem = gid & (WPACK_PER_DIR - 1);
  int k   = rem >> 10;                 // 0..511
  int n   = rem & (GDIM - 1);          // 0..1023
  const float* wih = d ? w_ih_b : w_ih_f;
  const float* whh = d ? w_hh_b : w_hh_f;
  float v = (k < D_) ? wih[n * D_ + k] : whh[n * H_ + (k - D_)];
  int kt = k >> 5, kin = k & 31;
  int nt = n >> 4, nin = n & 15;
  int lane = nin + 16 * ((kin >> 3) & 1);
  int slot = (kin & 7) + 8 * (kin >> 4);
  wpack[(size_t)d * WPACK_PER_DIR + ((nt * 16 + kt) * 512) + lane * 16 + slot] = (__bf16)v;
}

__global__ void zero_f32_kernel(float* __restrict__ p, int n) {
  int i = blockIdx.x * blockDim.x + threadIdx.x;
  if (i < n) p[i] = 0.0f;
}

// out[n, j] = nodeSum[n] / (2H)  for all j in [0, 2H)
__global__ void finalize_kernel(const float* __restrict__ nodeSum,
                                float* __restrict__ out) {
  int i = blockIdx.x * blockDim.x + threadIdx.x; // over N_*2H
  out[i] = nodeSum[i >> 9] * (1.0f / 512.0f);
}

// fast gate nonlinearities: v_exp_f32 + v_rcp_f32 (no IEEE divide chains)
__device__ __forceinline__ float frcp(float x)  { return __builtin_amdgcn_rcpf(x); }
__device__ __forceinline__ float fsig(float x)  { return frcp(1.0f + __expf(-x)); }
__device__ __forceinline__ float ftanh(float x) {
  // tanh(x) = 1 - 2/(1 + e^(2x))
  return __builtin_fmaf(-2.0f, frcp(1.0f + __expf(2.0f * x)), 1.0f);
}

// pack two f32 -> packed bf16 pair (one dword)
__device__ __forceinline__ unsigned int pk_bf16(float a, float b) {
  union { __bf16 h[2]; unsigned int u; } p;
  p.h[0] = (__bf16)a; p.h[1] = (__bf16)b;
  return p.u;
}

// ---------------------------------------------------------------------
// Fused bidirectional LSTM. One WG = 64 nodes, full sequence, both dirs.
// Per step: g[64,1024] = [x_t | h] (64x512, bf16 in LDS, A-fragment layout)
//           @ Wpack (512x1024, bf16, B-fragment layout) + bias, via
//           v_wmma_f32_16x16x32_bf16. Wave w owns hidden strip [16w,16w+16)
//           across all 4 gates -> c stays in f32 registers.
// A-fragment (16x32 MxK bf16): lane = m + 16*bit3(k), slot = (k&7) + 8*bit4(k)
//   => per (m, kt): k 0-7 / 8-15 / 16-23 / 24-31 are 4 contiguous 16B runs.
// ---------------------------------------------------------------------
__global__ __launch_bounds__(WG_THREADS)
void bilstm_wmma_kernel(const float* __restrict__ emb,
                        const float* __restrict__ b_f,
                        const float* __restrict__ b_b,
                        const long long* __restrict__ neigh_idx,
                        const __bf16* __restrict__ wpack,
                        float* __restrict__ nodeSum) {
  __shared__ __bf16 xA[NB * D_];   // 32 KB, A-fragment ordered (4 mt x 8 kt)
  __shared__ __bf16 hA[NB * H_];   // 32 KB, same layout

  const int tid  = threadIdx.x;
  const int lane = tid & 31;
  const int w    = tid >> 5;       // wave id 0..15 -> hidden strip
  const int nin  = lane & 15;      // N column within 16-wide tile
  const int half = lane >> 4;      // row-half of C/D tiles
  const int wgBase = blockIdx.x * NB;

  for (int dir = 0; dir < 2; ++dir) {
    const float*  bias = dir ? b_b : b_f;
    const __bf16* wp   = wpack + (size_t)dir * WPACK_PER_DIR;

    float biasv[4];
#pragma unroll
    for (int gi = 0; gi < 4; ++gi) biasv[gi] = bias[gi * H_ + w * 16 + nin];

    // h0 = 0 (vectorized: 2048 uint4 over 512 threads)
    {
      uint4 z = {0u, 0u, 0u, 0u};
      uint4* hz = (uint4*)hA;
#pragma unroll
      for (int i = 0; i < 4; ++i) hz[tid + i * WG_THREADS] = z;
    }

    float cReg[4][8];   // cell state: [m-tile][row], this lane's hidden column
    float sumReg[4][8]; // sum_t h
#pragma unroll
    for (int mt = 0; mt < 4; ++mt)
#pragma unroll
      for (int r = 0; r < 8; ++r) { cReg[mt][r] = 0.0f; sumReg[mt][r] = 0.0f; }

    __syncthreads();

    // precomputed h-writeback addressing (wave-constant / lane-constant)
    const int kt_h   = w >> 1;                       // hA fragment kt for this wave's strip
    const int slot_h = (nin & 7) + ((w & 1) << 3);
    const int half_h = (nin >> 3) & 1;

    for (int t = 0; t < S_; ++t) {
      const int s = dir ? (S_ - 1 - t) : t;

      // ---- gather x_t -> bf16 A-fragment layout in LDS (4x ds_store_b128) ----
      {
        const int nd = tid >> 3;           // node within block, 8 threads/node
        const int k0 = (tid & 7) * 32;     // 32 dims per thread == one kt fragment
        const long long row = neigh_idx[(long long)(wgBase + nd) * S_ + s];
        const float4* src = (const float4*)(emb + (long long)row * D_ + k0);
        const int mloc = nd & 15;
        const int fragbase = ((nd >> 4) * 8 + (tid & 7)) * 512;

        float v[32];
#pragma unroll
        for (int q = 0; q < 8; ++q) {
          float4 f = src[q];
          v[q * 4 + 0] = f.x; v[q * 4 + 1] = f.y; v[q * 4 + 2] = f.z; v[q * 4 + 3] = f.w;
        }
        uint4 seg[4];  // seg[j] = kins [8j, 8j+8) packed as 4 dwords
#pragma unroll
        for (int j = 0; j < 4; ++j) {
          seg[j].x = pk_bf16(v[j * 8 + 0], v[j * 8 + 1]);
          seg[j].y = pk_bf16(v[j * 8 + 2], v[j * 8 + 3]);
          seg[j].z = pk_bf16(v[j * 8 + 4], v[j * 8 + 5]);
          seg[j].w = pk_bf16(v[j * 8 + 6], v[j * 8 + 7]);
        }
        // kin 0-7 -> lane mloc,    slots 0-7 ; kin 8-15  -> lane mloc+16, slots 0-7
        // kin 16-23 -> lane mloc,  slots 8-15; kin 24-31 -> lane mloc+16, slots 8-15
        *(uint4*)&xA[fragbase + mloc * 16]            = seg[0];
        *(uint4*)&xA[fragbase + (mloc + 16) * 16]     = seg[1];
        *(uint4*)&xA[fragbase + mloc * 16 + 8]        = seg[2];
        *(uint4*)&xA[fragbase + (mloc + 16) * 16 + 8] = seg[3];
      }
      __syncthreads();   // xA ready; hA (prev step) ready

      // ---- fused GEMM: acc[gate][mt] (+bias) ----
      v8f acc[4][4];
#pragma unroll
      for (int gi = 0; gi < 4; ++gi)
#pragma unroll
        for (int mt = 0; mt < 4; ++mt) {
          v8f a;
#pragma unroll
          for (int r = 0; r < 8; ++r) a[r] = biasv[gi];
          acc[gi][mt] = a;
        }

      for (int kt = 0; kt < 16; ++kt) {
        const __bf16* Ab = (kt < 8) ? xA : hA;
        const int ktl = kt & 7;
        v16bf af[4];
#pragma unroll
        for (int mt = 0; mt < 4; ++mt)
          af[mt] = *reinterpret_cast<const v16bf*>(Ab + (mt * 8 + ktl) * 512 + lane * 16);
#pragma unroll
        for (int gi = 0; gi < 4; ++gi) {
          const int nt = gi * 16 + w;   // this wave's tile within gate gi
          v16bf bf = *reinterpret_cast<const v16bf*>(wp + (nt * 16 + kt) * 512 + lane * 16);
#pragma unroll
          for (int mt = 0; mt < 4; ++mt)
            acc[gi][mt] = __builtin_amdgcn_wmma_f32_16x16x32_bf16(
                false, af[mt], false, bf, (short)0, acc[gi][mt], false, false);
        }
      }
      __syncthreads();   // all LDS reads done -> safe to overwrite hA / xA

      // ---- gates, cell update, h writeback + time-sum ----
#pragma unroll
      for (int mt = 0; mt < 4; ++mt) {
#pragma unroll
        for (int r = 0; r < 8; ++r) {
          float g_i = acc[0][mt][r];
          float g_f = acc[1][mt][r];
          float g_g = acc[2][mt][r];
          float g_o = acc[3][mt][r];
          float c = __builtin_fmaf(fsig(g_f), cReg[mt][r], fsig(g_i) * ftanh(g_g));
          cReg[mt][r] = c;
          float h = fsig(g_o) * ftanh(c);
          sumReg[mt][r] += h;
          int m_in = r + 8 * half;     // row within 16-row tile
          hA[(mt * 8 + kt_h) * 512 + (m_in + 16 * half_h) * 16 + slot_h] = (__bf16)h;
        }
      }
      // hA writes become visible at next iteration's post-gather barrier
    }

    // ---- per-node reduction over this wave's 16 hidden columns ----
#pragma unroll
    for (int mt = 0; mt < 4; ++mt) {
#pragma unroll
      for (int r = 0; r < 8; ++r) {
        float sv = sumReg[mt][r];
        sv += __shfl_xor(sv, 1, 32);
        sv += __shfl_xor(sv, 2, 32);
        sv += __shfl_xor(sv, 4, 32);
        sv += __shfl_xor(sv, 8, 32);   // lanes 0-15 and 16-31 reduce separately
        if (nin == 0)
          atomicAdd(&nodeSum[wgBase + mt * 16 + r + 8 * half], sv);
      }
    }
    __syncthreads();  // before re-zeroing hA for the backward direction
  }
}

// ---------------------------------------------------------------------
extern "C" void kernel_launch(void* const* d_in, const int* in_sizes, int n_in,
                              void* d_out, int out_size, void* d_ws, size_t ws_size,
                              hipStream_t stream) {
  (void)in_sizes; (void)n_in; (void)out_size; (void)ws_size;
  const float*     emb    = (const float*)d_in[0];
  const float*     w_ih_f = (const float*)d_in[1];
  const float*     w_hh_f = (const float*)d_in[2];
  const float*     b_f    = (const float*)d_in[3];
  const float*     w_ih_b = (const float*)d_in[4];
  const float*     w_hh_b = (const float*)d_in[5];
  const float*     b_b    = (const float*)d_in[6];
  const long long* neigh  = (const long long*)d_in[7];

  __bf16* wpack   = (__bf16*)d_ws;                                   // 2 MB
  float*  nodeSum = (float*)((char*)d_ws + (size_t)2 * WPACK_PER_DIR * sizeof(__bf16));

  pack_weights_kernel<<<(2 * WPACK_PER_DIR + 255) / 256, 256, 0, stream>>>(
      w_ih_f, w_hh_f, w_ih_b, w_hh_b, wpack);
  zero_f32_kernel<<<(N_ + 255) / 256, 256, 0, stream>>>(nodeSum, N_);
  bilstm_wmma_kernel<<<N_ / NB, WG_THREADS, 0, stream>>>(
      emb, b_f, b_b, neigh, wpack, nodeSum);
  finalize_kernel<<<(N_ * 2 * H_) / 256, 256, 0, stream>>>(nodeSum, (float*)d_out);
}
